// Attention_10058813407378
// MI455X (gfx1250) — matrix operand order repaired and checked by reading, not compiled, not run on hardware
//
#include <hip/hip_runtime.h>
#include <hip/hip_bf16.h>

// ---------------------------------------------------------------------------
// Types for CDNA5 WMMA (wave32): v_wmma_f32_16x16x32_bf16
// ---------------------------------------------------------------------------
typedef __attribute__((ext_vector_type(16))) __bf16         v16bf;
typedef __attribute__((ext_vector_type(8)))  float          v8f;
typedef __attribute__((ext_vector_type(8)))  unsigned short ushort8;

union FragBF {
    v16bf   v;
    ushort8 h[2];
};

// ---------------------------------------------------------------------------
// Fragment loaders matching the CDNA5 ISA VGPR layouts (CDNA5 ISA §7.12.2)
//
// A (16x32 bf16, M x K): lane m = lane&15.
//   lanes 0-15 : VGPR0-3 = K 0..7,  VGPR4-7 = K 16..23
//   lanes 16-31: VGPR0-3 = K 8..15, VGPR4-7 = K 24..31
// B (32x16 bf16, K x N): lane n = lane&15.
//   lanes 0-15 : VGPR0-3 = K 0..7,  VGPR4-7 = K 16..23 of column n
//   lanes 16-31: VGPR0-3 = K 8..15, VGPR4-7 = K 24..31 of column n
// Works identically on global or LDS (generic) pointers.
// ---------------------------------------------------------------------------
__device__ __forceinline__ void load_frag_a(const __bf16* p, int ld,
                                            int lane, FragBF& f) {
    const int m   = lane & 15;
    const int sel = lane >> 4;                       // 0 or 1
    const __bf16* q = p + (size_t)m * ld + sel * 8;
    f.h[0] = *reinterpret_cast<const ushort8*>(q);
    f.h[1] = *reinterpret_cast<const ushort8*>(q + 16);
}

__device__ __forceinline__ void load_frag_b(const __bf16* p, int ld,
                                            int lane, FragBF& f) {
    const int n   = lane & 15;
    const int sel = lane >> 4;                       // 0 or 1
    const __bf16* q = p + (size_t)n * ld + sel * 8;
    f.h[0] = *reinterpret_cast<const ushort8*>(q);
    f.h[1] = *reinterpret_cast<const ushort8*>(q + 16);
}

// ---------------------------------------------------------------------------
// LDS byte-address for async-to-LDS instructions (addrspacecast -> low bits)
// ---------------------------------------------------------------------------
__device__ __forceinline__ uint32_t lds_addr_u32(const void* p) {
    return (uint32_t)(uintptr_t)(const __attribute__((address_space(3))) void*)p;
}

// ---------------------------------------------------------------------------
// Async DMA of one 128(rows) x 64(k) bf16 tile, global -> LDS (padded ld 72).
// 256 threads x 4 chunks x 16B = 16 KB. Tracked by ASYNCcnt.
// GVS addressing: mem = SGPR64(base) + VGPR32(byte offset)
// ---------------------------------------------------------------------------
__device__ __forceinline__ void async_copy_tile_128x64(
    const __bf16* g /* tile origin (row0,k0) */, int ldg,
    uint32_t lds_base, int t) {
#pragma unroll
    for (int i = 0; i < 4; ++i) {
        const int c   = t + 256 * i;
        const int row = c >> 3;                 // 0..127
        const int kc  = (c & 7) << 3;           // 0,8,..,56
        const uint32_t la = lds_base + (uint32_t)(row * 72 + kc) * 2u;
        const uint32_t go = (uint32_t)(row * ldg + kc) * 2u;
        asm volatile("global_load_async_to_lds_b128 %0, %1, %2"
                     :: "v"(la), "v"(go), "s"(g)
                     : "memory");
    }
}

__device__ __forceinline__ void wait_async_all() {
    asm volatile("s_wait_asynccnt 0" ::: "memory");
}

// ---------------------------------------------------------------------------
// NT GEMM core: C[m,n] = sum_k A[m,k]*B[n,k] (+ bias[n])
// A: M x K row-major bf16, B: N x K row-major bf16 (ld = K, K % 64 == 0)
// Block: 256 threads (8 waves), tile 128(M) x 128(N), K-step 64,
// double-buffered async LDS staging; wave tile 32(M) x 64(N).
// ---------------------------------------------------------------------------
template <bool HAS_BIAS, typename OutT>
__device__ __forceinline__ void gemm_nt_core(const __bf16* __restrict__ A,
                                             const __bf16* __restrict__ Bm,
                                             const float*  __restrict__ bias,
                                             OutT* __restrict__ C,
                                             int K, int N) {
    __shared__ __bf16 ldsA[2][128][72];
    __shared__ __bf16 ldsB[2][128][72];

    const int t    = threadIdx.x;
    const int lane = t & 31;
    const int wave = t >> 5;
    const int wm   = wave & 3;                 // 0..3 : M strip
    const int wn   = wave >> 2;                // 0..1 : N strip
    const int bm0  = blockIdx.y * 128;
    const int bn0  = blockIdx.x * 128;

    const __bf16* Ab = A  + (size_t)bm0 * K;
    const __bf16* Bb = Bm + (size_t)bn0 * K;

    const uint32_t aL[2] = { lds_addr_u32(&ldsA[0][0][0]), lds_addr_u32(&ldsA[1][0][0]) };
    const uint32_t bL[2] = { lds_addr_u32(&ldsB[0][0][0]), lds_addr_u32(&ldsB[1][0][0]) };

    v8f acc[2][4] = {};

    // prologue: DMA first K-block into buffer 0
    async_copy_tile_128x64(Ab, K, aL[0], t);
    async_copy_tile_128x64(Bb, K, bL[0], t);
    wait_async_all();
    __syncthreads();

    const int nkb = K >> 6;
    for (int kb = 0; kb < nkb; ++kb) {
        const int cur = kb & 1;
        // prefetch next K-block into the other buffer (overlaps with WMMAs)
        if (kb + 1 < nkb) {
            async_copy_tile_128x64(Ab + (kb + 1) * 64, K, aL[cur ^ 1], t);
            async_copy_tile_128x64(Bb + (kb + 1) * 64, K, bL[cur ^ 1], t);
        }
#pragma unroll
        for (int kk = 0; kk < 64; kk += 32) {
            FragBF a[2], b[4];
            load_frag_a(&ldsA[cur][wm * 32][kk],      72, lane, a[0]);
            load_frag_a(&ldsA[cur][wm * 32 + 16][kk], 72, lane, a[1]);
#pragma unroll
            for (int j = 0; j < 4; ++j)
                load_frag_b(&ldsB[cur][wn * 64 + 16 * j][kk], 72, lane, b[j]);
#pragma unroll
            for (int i = 0; i < 2; ++i)
#pragma unroll
                for (int j = 0; j < 4; ++j)
                    acc[i][j] = __builtin_amdgcn_wmma_f32_16x16x32_bf16(
                        false, a[i].v, false, b[j].v, (short)0, acc[i][j], false, false);
        }
        wait_async_all();
        __syncthreads();
    }

    // D layout: n = lane&15 ; m = v + 8*(lane>>4)
    const int nc = lane & 15;
    const int mh = (lane >> 4) * 8;
#pragma unroll
    for (int i = 0; i < 2; ++i)
#pragma unroll
        for (int j = 0; j < 4; ++j) {
            float bv = 0.0f;
            if (HAS_BIAS) bv = bias[bn0 + wn * 64 + j * 16 + nc];
            const int mrow = bm0 + wm * 32 + i * 16 + mh;
            const int ncol = bn0 + wn * 64 + j * 16 + nc;
#pragma unroll
            for (int v = 0; v < 8; ++v)
                C[(size_t)(mrow + v) * N + ncol] = (OutT)(acc[i][j][v] + bv);
        }
}

__global__ __launch_bounds__(256)
void gemm_nt_bias_bf16(const __bf16* __restrict__ A,
                       const __bf16* __restrict__ Bm,
                       const float*  __restrict__ bias,
                       __bf16* __restrict__ C, int K, int N) {
    gemm_nt_core<true, __bf16>(A, Bm, bias, C, K, N);
}

__global__ __launch_bounds__(256)
void gemm_nt_f32(const __bf16* __restrict__ A,
                 const __bf16* __restrict__ Bm,
                 float* __restrict__ C, int K, int N,
                 size_t strideA, size_t strideB, size_t strideC) {
    gemm_nt_core<false, float>(A  + (size_t)blockIdx.z * strideA,
                               Bm + (size_t)blockIdx.z * strideB,
                               nullptr,
                               C  + (size_t)blockIdx.z * strideC, K, N);
}

// ---------------------------------------------------------------------------
// f32 -> bf16 conversion (grid-stride)
// ---------------------------------------------------------------------------
__global__ void cvt_f32_to_bf16(const float* __restrict__ src,
                                __bf16* __restrict__ dst, int n) {
    int i      = blockIdx.x * blockDim.x + threadIdx.x;
    int stride = gridDim.x * blockDim.x;
    for (; i < n; i += stride) dst[i] = (__bf16)src[i];
}

// ---------------------------------------------------------------------------
// Row softmax over 1024 f32 columns, bf16 out. One block (256 thr) per row.
// ---------------------------------------------------------------------------
__global__ __launch_bounds__(256)
void softmax_rows_1024(const float* __restrict__ S, __bf16* __restrict__ out) {
    __shared__ float red[256];
    const size_t row = blockIdx.x;
    const int    t   = threadIdx.x;
    const float4 x   = reinterpret_cast<const float4*>(S + row * 1024)[t];

    float m = fmaxf(fmaxf(x.x, x.y), fmaxf(x.z, x.w));
    red[t] = m;
    __syncthreads();
    for (int s = 128; s > 0; s >>= 1) {
        if (t < s) red[t] = fmaxf(red[t], red[t + s]);
        __syncthreads();
    }
    m = red[0];
    __syncthreads();

    const float e0 = __expf(x.x - m), e1 = __expf(x.y - m);
    const float e2 = __expf(x.z - m), e3 = __expf(x.w - m);
    red[t] = e0 + e1 + e2 + e3;
    __syncthreads();
    for (int s = 128; s > 0; s >>= 1) {
        if (t < s) red[t] += red[t + s];
        __syncthreads();
    }
    const float inv = 1.0f / red[0];

    __bf16* o = out + row * 1024 + t * 4;
    o[0] = (__bf16)(e0 * inv);
    o[1] = (__bf16)(e1 * inv);
    o[2] = (__bf16)(e2 * inv);
    o[3] = (__bf16)(e3 * inv);
}

// ---------------------------------------------------------------------------
// Batched 1024x1024 bf16 transpose (32x32 LDS tiles, +1 pad vs bank conflicts)
// ---------------------------------------------------------------------------
__global__ __launch_bounds__(256)
void transpose_bf16_1024(const __bf16* __restrict__ in, __bf16* __restrict__ out) {
    __shared__ __bf16 tile[32][33];
    const size_t base = (size_t)blockIdx.z * 1048576;
    const int x0 = blockIdx.x * 32, y0 = blockIdx.y * 32;
    const int tx = threadIdx.x, ty = threadIdx.y;      // 32 x 8
#pragma unroll
    for (int i = 0; i < 32; i += 8)
        tile[ty + i][tx] = in[base + (size_t)(y0 + ty + i) * 1024 + x0 + tx];
    __syncthreads();
#pragma unroll
    for (int i = 0; i < 32; i += 8)
        out[base + (size_t)(x0 + ty + i) * 1024 + y0 + tx] = tile[tx][ty + i];
}

// ---------------------------------------------------------------------------
// Host launcher
// inputs: h[16,1024,1024]f32, s[16,1024,1024]f32, Wh[1024,1024]f32, bh[1024]f32,
//         Ws[1024,1024]f32, bs[1024]f32 ; out: [16*1024, 1024] f32
// ---------------------------------------------------------------------------
extern "C" void kernel_launch(void* const* d_in, const int* in_sizes, int n_in,
                              void* d_out, int out_size, void* d_ws, size_t ws_size,
                              hipStream_t stream) {
    const float* h  = (const float*)d_in[0];
    const float* s  = (const float*)d_in[1];
    const float* Wh = (const float*)d_in[2];
    const float* bh = (const float*)d_in[3];
    const float* Ws = (const float*)d_in[4];
    const float* bs = (const float*)d_in[5];

    const size_t ETOK = 16777216;   // 16*1024*1024 elements
    const size_t W1M  = 1048576;    // 1024*1024 elements

    uint8_t* ws = (uint8_t*)d_ws;
    // Region map (bytes), with lifetime-based reuse:
    //  [0, 64MB)       : hb16 (32MB) + sb16 (32MB)   -> later scores f32 (64MB)
    //                                                 -> later hpT bf16 (32MB)
    //  [64MB, 66MB)    : Whb16
    //  [66MB, 68MB)    : Wsb16
    //  [68MB, 100MB)   : hp16
    //  [100MB, 132MB)  : sp16                        -> later sm16
    __bf16* hb16   = (__bf16*)(ws + 0);
    __bf16* sb16   = hb16 + ETOK;
    float*  scores = (float*)(ws + 0);
    __bf16* hpT16  = (__bf16*)(ws + 0);
    __bf16* Whb16  = (__bf16*)(ws + 67108864);
    __bf16* Wsb16  = (__bf16*)(ws + 69206016);
    __bf16* hp16   = (__bf16*)(ws + 71303168);
    __bf16* sp16   = (__bf16*)(ws + 104857600);
    __bf16* sm16   = sp16;   // reuse after scores are materialized

    // 1) quantize inputs/weights to bf16
    cvt_f32_to_bf16<<<4096, 256, 0, stream>>>(h,  hb16,  (int)ETOK);
    cvt_f32_to_bf16<<<4096, 256, 0, stream>>>(s,  sb16,  (int)ETOK);
    cvt_f32_to_bf16<<<1024, 256, 0, stream>>>(Wh, Whb16, (int)W1M);
    cvt_f32_to_bf16<<<1024, 256, 0, stream>>>(Ws, Wsb16, (int)W1M);

    // 2) projections: hp = h@Wh^T + bh ; sp = s@Ws^T + bs   (M=16384, N=K=1024)
    gemm_nt_bias_bf16<<<dim3(8, 128, 1), 256, 0, stream>>>(hb16, Whb16, bh, hp16, 1024, 1024);
    gemm_nt_bias_bf16<<<dim3(8, 128, 1), 256, 0, stream>>>(sb16, Wsb16, bs, sp16, 1024, 1024);

    // 3) scores[b] = hp[b] @ sp[b]^T  (f32, overwrites dead hb16/sb16 region)
    gemm_nt_f32<<<dim3(8, 8, 16), 256, 0, stream>>>(hp16, sp16, scores,
                                                    1024, 1024, W1M, W1M, W1M);

    // 4) row softmax -> bf16 sm (overwrites dead sp16 region)
    softmax_rows_1024<<<16384, 256, 0, stream>>>(scores, sm16);

    // 5) hpT[b] = hp[b]^T  (overwrites dead scores region)
    transpose_bf16_1024<<<dim3(32, 32, 16), dim3(32, 8), 0, stream>>>(hp16, hpT16);

    // 6) res[b] = sm[b] @ hpT[b]^T  -> d_out (f32)
    gemm_nt_f32<<<dim3(8, 8, 16), 256, 0, stream>>>(sm16, hpT16, (float*)d_out,
                                                    1024, 1024, W1M, W1M, W1M);
}
